// FenwickTree_31301721653836
// MI455X (gfx1250) — matrix operand order repaired and checked by reading, not compiled, not run on hardware
//
#include <hip/hip_runtime.h>

// ---------------------------------------------------------------------------
// Fenwick-forest TreeLSTM on MI455X (gfx1250).
//   D = 128, gates = 5*D = 640, cell GEMM: (B,256) @ (256,640) via
//   v_wmma_f32_16x16x32_f16 (f16 in, f32 acc), K = 8 chunks of 32.
//   M-tile = 32 rows/block (2 sub-tiles share every B fragment -> halves
//   L2 weight traffic). Stride-8 n-tile ownership: wave w owns gate-cols
//   [16w,16w+16) of ALL 5 gates, so LSTM gating runs fully in registers.
// ---------------------------------------------------------------------------

#define D_       128
#define GATES    640          // 5*D
#define KDIM     256          // 2*D
#define NCHUNKS  8            // KDIM / 32
#define NTILES   40           // GATES / 16
#define TPB      256          // 8 wave32
#define MROWS    32           // output rows per block (2 WMMA row-tiles)

typedef __attribute__((ext_vector_type(16))) _Float16 v16h;
typedef __attribute__((ext_vector_type(8)))  float    v8f;

__device__ __forceinline__ float sigm(float x) { return 1.0f / (1.0f + __expf(-x)); }

// ---------------------------------------------------------------------------
// Segment bookkeeping. At level t the state buffer holds, in order k=16..0,
// segment k with s_k(t) = (k>t) ? 2^(k-t) : 1 rows. Output row R of level t
// is either a merge of two input rows or a pass-through copy.
// ---------------------------------------------------------------------------
__device__ void map_row(int R, int t, int* srcL, int* srcR, int* pass) {
  int off_out = 0, off_in = 0;
  for (int k = 16; k >= 0; --k) {
    int so = (k > t + 1) ? (1 << (k - t - 1)) : 1;
    int si = (k > t)     ? (1 << (k - t))     : 1;
    if (R < off_out + so) {
      int idx = R - off_out;
      if (si > 1) { *pass = 0; *srcL = off_in + 2 * idx; *srcR = *srcL + 1; }
      else        { *pass = 1; *srcL = off_in; *srcR = off_in; }
      return;
    }
    off_out += so; off_in += si;
  }
  *pass = 1; *srcL = 0; *srcR = 0;
}

// ---------------------------------------------------------------------------
// Weight swizzle: Wcat(256x640) f32 -> f16 fragments; each lane's B fragment
// for (ntile, kchunk) is 16 contiguous halves (one 32B global load).
// B 32x16 f16 layout used here (contiguous k, matches A; not ISA table order): lanes 0-15 hold K=0..15 (N = lane),
// lanes 16-31 hold K=16..31; half i -> K = (lane>>4)*16 + i.
// ---------------------------------------------------------------------------
__global__ __launch_bounds__(TPB)
void swizzle_weights_kernel(const float* __restrict__ Wl,
                            const float* __restrict__ Wr,
                            _Float16* __restrict__ dst) {
  int x = blockIdx.x * TPB + threadIdx.x;     // 0 .. 40*8*32*16-1
  int i     = x & 15;
  int lane  = (x >> 4) & 31;
  int chunk = (x >> 9) & 7;
  int ntile = x >> 12;
  int hi  = lane >> 4;
  int k   = chunk * 32 + hi * 16 + i;
  int n   = ntile * 16 + (lane & 15);
  float v = (k < D_) ? Wl[(size_t)k * GATES + n]
                     : Wr[(size_t)(k - D_) * GATES + n];
  dst[x] = (_Float16)v;
}

// ---------------------------------------------------------------------------
// A fragment mapping (16x32 f16, contiguous k as in B, not ISA table order): lane = row + 16*hi where
// hi=k32>>4, half i=k32&15.
// ---------------------------------------------------------------------------
__device__ __forceinline__ void stage_a_half(_Float16* lds_a, int r16, int kk, float v) {
  int chunk = kk >> 5, k32 = kk & 31;
  int hi = k32 >> 4;
  int i  = k32 & 15;
  int lane = r16 + (hi << 4);
  lds_a[(chunk * 32 + lane) * 16 + i] = (_Float16)v;
}

// B fragment pointer for (gate n, wave w, chunk): ntile = n*8 + w.
__device__ __forceinline__ const v16h* b_frag(const _Float16* __restrict__ Wswz,
                                              int n, int wave, int chunk, int lane) {
  return (const v16h*)&Wswz[(size_t)(((n * 8 + wave) * NCHUNKS + chunk) * 32 + lane) * 16];
}

// ---------------------------------------------------------------------------
// One tree level: each block produces 32 parent rows. Wave w accumulates all
// 5 gates for cols [16w,16w+16) of both 16-row sub-tiles, then gates in-reg.
// ---------------------------------------------------------------------------
__global__ __launch_bounds__(TPB)
void tree_level_kernel(const float* __restrict__ in_h, const float* __restrict__ in_c,
                       const _Float16* __restrict__ Wswz, const float* __restrict__ bias,
                       float* __restrict__ out_h, float* __restrict__ out_c,
                       int t, int rowsOut) {
  __shared__ __align__(32) _Float16 lds_a[2 * NCHUNKS * 32 * 16];   // 16 KB
  __shared__ int sL[MROWS], sR[MROWS], sP[MROWS];

  const int tid = threadIdx.x;
  const int baseRow = blockIdx.x * MROWS;

  if (tid < MROWS) {
    int R = baseRow + tid;
    if (R >= rowsOut) R = rowsOut - 1;
    map_row(R, t, &sL[tid], &sR[tid], &sP[tid]);
  }
  __syncthreads();

  // Stage A: 32 rows x 256 K halves (f32 state -> f16 fragment order).
#pragma unroll
  for (int e = 0; e < 32; ++e) {
    int x = e * TPB + tid;                 // 0..8191
    int r = x >> 8, kk = x & 255;
    float v = (kk < D_) ? in_h[(size_t)sL[r] * D_ + kk]
                        : in_h[(size_t)sR[r] * D_ + (kk - D_)];
    stage_a_half(&lds_a[(r >> 4) * NCHUNKS * 32 * 16], r & 15, kk, v);
  }
  __syncthreads();

  const int wave = tid >> 5, lane = tid & 31;
  const int hi = lane >> 4, n16 = lane & 15;

  v8f acc[2][5];
#pragma unroll
  for (int s = 0; s < 2; ++s)
#pragma unroll
    for (int n = 0; n < 5; ++n) acc[s][n] = (v8f){0.f,0.f,0.f,0.f,0.f,0.f,0.f,0.f};

#pragma unroll
  for (int chunk = 0; chunk < NCHUNKS; ++chunk) {
    v16h a0 = *(const v16h*)&lds_a[(chunk * 32 + lane) * 16];
    v16h a1 = *(const v16h*)&lds_a[(NCHUNKS * 32 + chunk * 32 + lane) * 16];
#pragma unroll
    for (int n = 0; n < 5; ++n) {
      v16h b = *b_frag(Wswz, n, wave, chunk, lane);
      acc[0][n] = __builtin_amdgcn_wmma_f32_16x16x32_f16(
          false, a0, false, b, (short)0, acc[0][n], false, false);
      acc[1][n] = __builtin_amdgcn_wmma_f32_16x16x32_f16(
          false, a1, false, b, (short)0, acc[1][n], false, false);
    }
  }

  // In-register gating. Lane owns col c for 8 rows per sub-tile (M = v+8*hi).
  const int c = wave * 16 + n16;
  const float b_i = bias[          c];
  const float b_o = bias[1 * D_ + c];
  const float b_u = bias[2 * D_ + c];
  const float b_l = bias[3 * D_ + c];
  const float b_r = bias[4 * D_ + c];

#pragma unroll
  for (int s = 0; s < 2; ++s) {
#pragma unroll
    for (int v = 0; v < 8; ++v) {
      int M = s * 16 + v + 8 * hi;
      int R = baseRow + M;
      if (R >= rowsOut) continue;
      float h, cc;
      if (sP[M]) {                         // finished segment: carry root
        h  = in_h[(size_t)sL[M] * D_ + c];
        cc = in_c[(size_t)sL[M] * D_ + c];
      } else {
        float cl = in_c[(size_t)sL[M] * D_ + c];
        float cr = in_c[(size_t)sR[M] * D_ + c];
        float gi = acc[s][0][v] + b_i;
        float go = acc[s][1][v] + b_o;
        float gu = acc[s][2][v] + b_u;
        float gl = acc[s][3][v] + b_l;
        float gr = acc[s][4][v] + b_r;
        cc = sigm(gi) * tanhf(gu) + sigm(gl) * cl + sigm(gr) * cr;
        h  = sigm(go) * tanhf(cc);
      }
      out_h[(size_t)R * D_ + c] = h;
      out_c[(size_t)R * D_ + c] = cc;
    }
  }
}

// ---------------------------------------------------------------------------
// One chain cell (batch 1): A row 0 = [prev_h | root_h], rows 1..15 zero.
// Row 0 of the D tile lives in lanes 0-15, VGPR 0 -> gate in registers.
// ---------------------------------------------------------------------------
__global__ __launch_bounds__(TPB)
void chain_cell_kernel(const float* __restrict__ prev_h, const float* __restrict__ prev_c,
                       const float* __restrict__ root_h, const float* __restrict__ root_c,
                       const _Float16* __restrict__ Wswz, const float* __restrict__ bias,
                       float* __restrict__ out_h, float* __restrict__ out_c) {
  __shared__ __align__(32) _Float16 lds_a[NCHUNKS * 32 * 16];   // 8 KB
  const int tid = threadIdx.x;

#pragma unroll
  for (int e = 0; e < 16; ++e) {
    int x = e * TPB + tid;
    int r = x >> 8, kk = x & 255;
    float v = 0.0f;
    if (r == 0) v = (kk < D_) ? prev_h[kk] : root_h[kk - D_];
    stage_a_half(lds_a, r, kk, v);
  }
  __syncthreads();

  const int wave = tid >> 5, lane = tid & 31;
  v8f acc[5];
#pragma unroll
  for (int n = 0; n < 5; ++n) acc[n] = (v8f){0.f,0.f,0.f,0.f,0.f,0.f,0.f,0.f};

#pragma unroll
  for (int chunk = 0; chunk < NCHUNKS; ++chunk) {
    v16h a = *(const v16h*)&lds_a[(chunk * 32 + lane) * 16];
#pragma unroll
    for (int n = 0; n < 5; ++n) {
      v16h b = *b_frag(Wswz, n, wave, chunk, lane);
      acc[n] = __builtin_amdgcn_wmma_f32_16x16x32_f16(
          false, a, false, b, (short)0, acc[n], false, false);
    }
  }

  if (lane < 16) {                          // row M=0 -> lanes 0-15, VGPR 0
    int c = wave * 16 + lane;
    float gi = acc[0][0] + bias[          c];
    float go = acc[1][0] + bias[1 * D_ + c];
    float gu = acc[2][0] + bias[2 * D_ + c];
    float gl = acc[3][0] + bias[3 * D_ + c];
    float gr = acc[4][0] + bias[4 * D_ + c];
    float cc = sigm(gi) * tanhf(gu) + sigm(gl) * prev_c[c] + sigm(gr) * root_c[c];
    float h  = sigm(go) * tanhf(cc);
    out_h[c] = h;
    out_c[c] = cc;
  }
}

// ---------------------------------------------------------------------------
// Host orchestration (graph-capture safe: only async launches on `stream`).
// Workspace (~101 MB): swizzled merge/sum weights (2 x 320 KB), bufA h/c
// (65536 rows), bufB h/c (32769 rows), chain state s0/s1.
// ---------------------------------------------------------------------------
extern "C" void kernel_launch(void* const* d_in, const int* in_sizes, int n_in,
                              void* d_out, int out_size, void* d_ws, size_t ws_size,
                              hipStream_t stream) {
  (void)in_sizes; (void)n_in; (void)out_size; (void)ws_size;

  const float* h_leaves = (const float*)d_in[0];
  const float* c_leaves = (const float*)d_in[1];
  const float* Wl_m = (const float*)d_in[2];
  const float* Wr_m = (const float*)d_in[3];
  const float* b_m  = (const float*)d_in[4];
  const float* Wl_s = (const float*)d_in[5];
  const float* Wr_s = (const float*)d_in[6];
  const float* b_s  = (const float*)d_in[7];
  float* out = (float*)d_out;

  const size_t WBYTES = (size_t)KDIM * GATES * sizeof(_Float16);   // 327680
  const size_t ROWS_A = 65536;   // largest even-level output
  const size_t ROWS_B = 32769;   // largest odd-level output

  char* ws = (char*)d_ws;
  _Float16* Wm  = (_Float16*)ws;
  _Float16* Wsu = (_Float16*)(ws + WBYTES);
  float* bufA_h = (float*)(ws + 2 * WBYTES);
  float* bufA_c = bufA_h + ROWS_A * D_;
  float* bufB_h = bufA_c + ROWS_A * D_;
  float* bufB_c = bufB_h + ROWS_B * D_;
  float* s0_h   = bufB_c + ROWS_B * D_;
  float* s0_c   = s0_h + D_;
  float* s1_h   = s0_c + D_;
  float* s1_c   = s1_h + D_;

  // Swizzle both weight sets into WMMA B-fragment order (f32 -> f16).
  swizzle_weights_kernel<<<NTILES * NCHUNKS * 32 * 16 / TPB, TPB, 0, stream>>>(Wl_m, Wr_m, Wm);
  swizzle_weights_kernel<<<NTILES * NCHUNKS * 32 * 16 / TPB, TPB, 0, stream>>>(Wl_s, Wr_s, Wsu);

  // 16 tree levels; unified batch over all 17 segments, ping-pong state.
  const float* in_h = h_leaves;
  const float* in_c = c_leaves;
  for (int t = 0; t < 16; ++t) {
    int rowsOut = (1 << (16 - t)) + t;        // R(t+1) = 2^(16-t) + t
    float* oh = (t & 1) ? bufB_h : bufA_h;
    float* oc = (t & 1) ? bufB_c : bufA_c;
    int blocks = (rowsOut + MROWS - 1) / MROWS;
    tree_level_kernel<<<blocks, TPB, 0, stream>>>(in_h, in_c, Wm, b_m, oh, oc, t, rowsOut);
    in_h = oh; in_c = oc;
  }
  // After t=15 (odd), the 17 roots sit in bufB rows 0..16 (k=16 .. k=0).

  // Sequential sum-cell chain: start from k=0 root (row 16), fold rows 15..0.
  const float* ph = bufB_h + 16 * D_;
  const float* pc = bufB_c + 16 * D_;
  for (int j = 0; j < 16; ++j) {
    int rootRow = 15 - j;
    float *oh, *oc;
    if (j == 15)      { oh = out;  oc = out + D_; }   // final: [h | c] -> d_out
    else if (j & 1)   { oh = s1_h; oc = s1_c; }
    else              { oh = s0_h; oc = s0_c; }
    chain_cell_kernel<<<1, TPB, 0, stream>>>(ph, pc,
                                             bufB_h + (size_t)rootRow * D_,
                                             bufB_c + (size_t)rootRow * D_,
                                             Wsu, b_s, oh, oc);
    ph = oh; pc = oc;
  }
}
